// ResRnn_14345190769021
// MI455X (gfx1250) — matrix operand order repaired and checked by reading, not compiled, not run on hardware
//
#include <hip/hip_runtime.h>

// ---- CDNA5 (gfx1250) wave32 WMMA types ----
typedef __attribute__((ext_vector_type(16))) __bf16 v16bf;
typedef __attribute__((ext_vector_type(8)))  __bf16 v8bf;
typedef __attribute__((ext_vector_type(8)))  float  v8f;

#define LIN       0.99999f
#define SW        1024
#define BATCH     256
#define IW        64
#define SEQ       512
#define NBLK      32
#define KCH       64                  // K staged per LDS buffer
#define NIT       (SW / KCH)          // 16 iterations
#define LDS_PITCH 144                 // 128B row + 16B pad: 36-bank stride, 16B aligned
#define TILE_B    (64 * LDS_PITCH)    // one double-buffer slot (9216 B)

static __device__ __forceinline__ v16bf combine16(v8bf lo, v8bf hi) {
  return __builtin_shufflevector(lo, hi, 0,1,2,3,4,5,6,7,8,9,10,11,12,13,14,15);
}
static __device__ __forceinline__ v8bf cvt8(v8f v) {
  v8bf r;
#pragma unroll
  for (int i = 0; i < 8; ++i) r[i] = (__bf16)v[i];
  return r;
}
// CDNA5 async global->LDS copy (ASYNCcnt-tracked), 16 bytes per lane
static __device__ __forceinline__ void async_b128(unsigned ldsOff, const void* g) {
  asm volatile("global_load_async_to_lds_b128 %0, %1, off"
               :: "v"(ldsOff), "v"(g) : "memory");
}
static __device__ __forceinline__ void wait_async2(bool keep_two) {
  if (keep_two) asm volatile("s_wait_asynccnt 0x2" ::: "memory");
  else          asm volatile("s_wait_asynccnt 0x0" ::: "memory");
}

// device-wide software barrier: all NBLK blocks (guaranteed co-resident)
static __device__ __forceinline__ void grid_barrier(unsigned* ctr, unsigned& target) {
  __threadfence();                // device-scope release of this thread's stores
  __syncthreads();                // whole block arrived + fenced
  target += NBLK;
  if (threadIdx.x == 0) {
    __hip_atomic_fetch_add(ctr, 1u, __ATOMIC_RELEASE, __HIP_MEMORY_SCOPE_AGENT);
    while (__hip_atomic_load(ctr, __ATOMIC_ACQUIRE, __HIP_MEMORY_SCOPE_AGENT) < target)
      __builtin_amdgcn_s_sleep(1);
  }
  __syncthreads();                // acquire (incl. WGP$ inv) done for whole block
}

// ---------- one-time prep kernels ----------
__global__ __launch_bounds__(256) void cvt_w_bf16(const float* __restrict__ src,
                                                  __bf16* __restrict__ dst, int n) {
  int i = blockIdx.x * 256 + threadIdx.x;
  if (i < n) dst[i] = (__bf16)src[i];
}
__global__ __launch_bounds__(256) void init_stream0(const float* __restrict__ init,
                                                    float* __restrict__ s0) {
  int i = blockIdx.x * 256 + threadIdx.x;      // over BATCH*SW
  int c = i & (SW - 1);
  s0[i] = (1.0f - LIN) * init[c];
}
__global__ void zero_ctr(unsigned* ctr) { if (threadIdx.x == 0) *ctr = 0u; }

// ---------- shared GEMM machinery ----------
// Block b: nBase = (b&15)*64 (8 waves share one 64-col W group), mBase per wave.
// Cooperative stage: 64 rows x 128B (64x64 bf16 W-tile) per buffer; thread copies
// rows r and r+32 at chunk (tid&7): 2 x async b128.

static __device__ __forceinline__ void gemm1_phase(
    const float* __restrict__ x, const float* __restrict__ prev,
    const __bf16* __restrict__ W, const float* __restrict__ bias,
    __bf16* __restrict__ H, char* lds) {
  const int tid    = threadIdx.x;
  const int lane   = tid & 31;
  const int laneLo = lane & 15, laneHi = lane >> 4;
  const int nBase  = ((int)blockIdx.x & 15) << 6;
  const int mBase  = (((((int)blockIdx.x) >> 4) << 3) | (tid >> 5)) << 4;

  const int row0 = tid >> 3, ci = tid & 7;
  const __bf16*  gW0     = W + (size_t)(nBase + row0) * SW + ci * 8;
  const __bf16*  gW1     = gW0 + (size_t)32 * SW;
  const unsigned ldsBase = (unsigned)(uintptr_t)lds;
  const unsigned lo0     = row0 * LDS_PITCH + ci * 16;
  const unsigned lo1     = lo0 + 32 * LDS_PITCH;

  const int m = mBase + laneLo;
  const float* xRow = x    + (size_t)m * IW;
  const float* pRow = prev + (size_t)m * SW;
  const int kh = 8 * laneHi;

  v8f acc[4] = {};
  async_b128(ldsBase + lo0, gW0);
  async_b128(ldsBase + lo1, gW1);

#pragma unroll 2
  for (int i = 0; i < NIT; ++i) {
    const int      ko     = i * KCH;
    const unsigned curOff = (i & 1) ? (unsigned)TILE_B : 0u;
    const unsigned nxtOff = curOff ^ (unsigned)TILE_B;
    const bool more = (i + 1) < NIT;
    if (more) {
      async_b128(ldsBase + nxtOff + lo0, gW0 + ko + KCH);
      async_b128(ldsBase + nxtOff + lo1, gW1 + ko + KCH);
    }
    // A fragments (2 x K=32): fused shift (x | prev) + fp32->bf16
    v16bf a[2];
#pragma unroll
    for (int kk = 0; kk < 2; ++kk) {
      const int kc = ko + kk * 32 + kh;
      v8f flo = *(const v8f*)((kc      < IW) ? (xRow + kc)      : (pRow + kc      - IW));
      v8f fhi = *(const v8f*)((kc + 16 < IW) ? (xRow + kc + 16) : (pRow + kc + 16 - IW));
      a[kk] = combine16(cvt8(flo), cvt8(fhi));
    }
    wait_async2(more);           // retire current buffer's 2 copies (in-order)
    __syncthreads();             // all waves' copies visible
    v16bf bfrag[2][4];
    const char* base = lds + curOff;
#pragma unroll
    for (int kk = 0; kk < 2; ++kk)
#pragma unroll
      for (int n = 0; n < 4; ++n) {
        const char* p = base + (n * 16 + laneLo) * LDS_PITCH + kk * 64 + 16 * laneHi;
        bfrag[kk][n] = combine16(*(const v8bf*)p, *(const v8bf*)(p + 32));
      }
    __syncthreads();             // frags in regs; buffer reusable next iter
#pragma unroll
    for (int kk = 0; kk < 2; ++kk)
#pragma unroll
      for (int n = 0; n < 4; ++n)
        acc[n] = __builtin_amdgcn_wmma_f32_16x16x32_bf16(
            false, a[kk], false, bfrag[kk][n], (short)0, acc[n], false, false);
  }

  const int mRow = mBase + 8 * laneHi;
#pragma unroll
  for (int n = 0; n < 4; ++n) {
    const int   c  = nBase + n * 16 + laneLo;
    const float bc = bias[c];
#pragma unroll
    for (int r = 0; r < 8; ++r) {
      float v = acc[n][r] + bc;
      v = v > 0.0f ? v : 0.0f;
      H[(size_t)(mRow + r) * SW + c] = (__bf16)v;
    }
  }
}

static __device__ __forceinline__ void gemm2_phase(
    const __bf16* __restrict__ A, const __bf16* __restrict__ W,
    const float* __restrict__ bias, const float* __restrict__ x,
    const float* __restrict__ prev, float* __restrict__ out, char* lds) {
  const int tid    = threadIdx.x;
  const int lane   = tid & 31;
  const int laneLo = lane & 15, laneHi = lane >> 4;
  const int nBase  = ((int)blockIdx.x & 15) << 6;
  const int mBase  = (((((int)blockIdx.x) >> 4) << 3) | (tid >> 5)) << 4;

  const int row0 = tid >> 3, ci = tid & 7;
  const __bf16*  gW0     = W + (size_t)(nBase + row0) * SW + ci * 8;
  const __bf16*  gW1     = gW0 + (size_t)32 * SW;
  const unsigned ldsBase = (unsigned)(uintptr_t)lds;
  const unsigned lo0     = row0 * LDS_PITCH + ci * 16;
  const unsigned lo1     = lo0 + 32 * LDS_PITCH;

  const int kh = 8 * laneHi;
  const __bf16* aRow = A + (size_t)(mBase + laneLo) * SW + kh;

  v8f acc[4] = {};
  async_b128(ldsBase + lo0, gW0);
  async_b128(ldsBase + lo1, gW1);

#pragma unroll 2
  for (int i = 0; i < NIT; ++i) {
    const int      ko     = i * KCH;
    const unsigned curOff = (i & 1) ? (unsigned)TILE_B : 0u;
    const unsigned nxtOff = curOff ^ (unsigned)TILE_B;
    const bool more = (i + 1) < NIT;
    if (more) {
      async_b128(ldsBase + nxtOff + lo0, gW0 + ko + KCH);
      async_b128(ldsBase + nxtOff + lo1, gW1 + ko + KCH);
    }
    v16bf a[2];
#pragma unroll
    for (int kk = 0; kk < 2; ++kk) {
      const __bf16* p = aRow + ko + kk * 32;
      a[kk] = combine16(*(const v8bf*)p, *(const v8bf*)(p + 16));
    }
    wait_async2(more);
    __syncthreads();
    v16bf bfrag[2][4];
    const char* base = lds + curOff;
#pragma unroll
    for (int kk = 0; kk < 2; ++kk)
#pragma unroll
      for (int n = 0; n < 4; ++n) {
        const char* p = base + (n * 16 + laneLo) * LDS_PITCH + kk * 64 + 16 * laneHi;
        bfrag[kk][n] = combine16(*(const v8bf*)p, *(const v8bf*)(p + 32));
      }
    __syncthreads();
#pragma unroll
    for (int kk = 0; kk < 2; ++kk)
#pragma unroll
      for (int n = 0; n < 4; ++n)
        acc[n] = __builtin_amdgcn_wmma_f32_16x16x32_bf16(
            false, a[kk], false, bfrag[kk][n], (short)0, acc[n], false, false);
  }

  const int mRow = mBase + 8 * laneHi;
#pragma unroll
  for (int n = 0; n < 4; ++n) {
    const int   c  = nBase + n * 16 + laneLo;
    const float bc = bias[c];
#pragma unroll
    for (int r = 0; r < 8; ++r) {
      const int m = mRow + r;
      float y = acc[n][r] + bc;
      float s = (c < IW) ? x[m * IW + c] : prev[(size_t)m * SW + (c - IW)];
      out[(size_t)m * SW + c] = LIN * s + (1.0f - LIN) * y;
    }
  }
}

// ---------- persistent kernel: the whole 512-step recurrence ----------
__global__ __launch_bounds__(256) void resrnn_persistent(
    const float* __restrict__ inp, const float* __restrict__ stream0,
    const __bf16* __restrict__ W1bf, const float* __restrict__ b1,
    const __bf16* __restrict__ W2bf, const float* __restrict__ b2,
    __bf16* __restrict__ H, float* __restrict__ out, unsigned* __restrict__ ctr) {
  __shared__ alignas(16) char lds[2 * TILE_B];
  unsigned target = 0;
  for (int t = 0; t < SEQ; ++t) {
    const float* xt    = inp + (size_t)t * BATCH * IW;
    const float* prev  = t ? (out + (size_t)(t - 1) * BATCH * SW) : stream0;
    float*       out_t = out + (size_t)t * BATCH * SW;

    gemm1_phase(xt, prev, W1bf, b1, H, lds);
    grid_barrier(ctr, target);                 // H complete, visible device-wide
    gemm2_phase(H, W2bf, b2, xt, prev, out_t, lds);
    grid_barrier(ctr, target);                 // out_t complete -> prev of t+1
  }
}

extern "C" void kernel_launch(void* const* d_in, const int* in_sizes, int n_in,
                              void* d_out, int out_size, void* d_ws, size_t ws_size,
                              hipStream_t stream) {
  const float* inp = (const float*)d_in[0];  // (S, B, IW)
  const float* ini = (const float*)d_in[1];  // (SW,)
  const float* W1  = (const float*)d_in[2];  // (SW, SW)
  const float* b1  = (const float*)d_in[3];
  const float* W2  = (const float*)d_in[4];
  const float* b2  = (const float*)d_in[5];
  float* out = (float*)d_out;                // (S, B, SW)

  // workspace layout
  char* ws = (char*)d_ws;
  __bf16*   W1bf    = (__bf16*)(ws);                                  // 2 MB
  __bf16*   W2bf    = (__bf16*)(ws + (size_t)SW * SW * 2);            // 2 MB
  float*    stream0 = (float*)(ws + (size_t)SW * SW * 4);             // 1 MB
  __bf16*   hbf     = (__bf16*)(ws + (size_t)SW * SW * 4 + (size_t)BATCH * SW * 4);
  unsigned* ctr     = (unsigned*)(ws + (size_t)SW * SW * 4 + (size_t)BATCH * SW * 4
                                     + (size_t)BATCH * SW * 2);

  cvt_w_bf16<<<(SW * SW) / 256, 256, 0, stream>>>(W1, W1bf, SW * SW);
  cvt_w_bf16<<<(SW * SW) / 256, 256, 0, stream>>>(W2, W2bf, SW * SW);
  init_stream0<<<(BATCH * SW) / 256, 256, 0, stream>>>(ini, stream0);
  zero_ctr<<<1, 32, 0, stream>>>(ctr);

  resrnn_persistent<<<NBLK, 256, 0, stream>>>(inp, stream0, W1bf, b1, W2bf, b2,
                                              hbf, out, ctr);
}
